// ReconstToy_13881334301183
// MI455X (gfx1250) — matrix operand order repaired and checked by reading, not compiled, not run on hardware
//
#include <hip/hip_runtime.h>
#include <hip/hip_bf16.h>

typedef __attribute__((ext_vector_type(16))) _Float16 v16h;
typedef __attribute__((ext_vector_type(8)))  _Float16 v8h;
typedef __attribute__((ext_vector_type(8)))  float    v8f;

#define B_       8
#define T_       2048
#define F_       128
#define NT_      256      // num tatums
#define TILE     4        // output t-rows per block
#define NTHREADS 512      // 16 waves (wave32) -> 4 waves/SIMD32 to hide LDS latency

// ---------------- LDS layout (bytes, all offsets 32B aligned) ----------------
// a1 rows: t0-3 .. t0+6 (10), a2 rows: t0-2 .. t0+5 (8), a3 rows: t0-1 .. t0+4 (6)
// x0 rows: t0-4 .. t0+7 (12).  Columns p=0..129 map to f=-1..128 (zero halo).
#define OFF_A1 0
#define SZ_A1  (10*130*32*2)            // 83200
#define OFF_A3 0                         // alias: a1 is dead before a3 is written
#define OFF_A2 (OFF_A1 + SZ_A1)
#define SZ_A2  (8*130*32*2)             // 66560
#define OFF_W2 (OFF_A2 + SZ_A2)         // pre-swizzled WMMA A-fragments
#define SZ_WF  (9*2*32*16*2)            // 18432: [tap][mtile][lane][16 halves]
#define OFF_W3 (OFF_W2 + SZ_WF)
#define OFF_X0 (OFF_W3 + SZ_WF)
#define SZ_X0  (12*130*2*2)             // 6240
#define OFF_W1 (OFF_X0 + SZ_X0)
#define SZ_W1  (32*2*9*4)               // 2304
#define OFF_W4 (OFF_W1 + SZ_W1)
#define SZ_W4  (32*9*4)                 // 1152
#define SMEM_BYTES (OFF_W4 + SZ_W4)     // 196320 B  (< 320KB CDNA5 WGP LDS)

// One 3x3 conv layer 32ch -> 32ch as implicit GEMM on v_wmma_f32_16x16x32_f16.
// Each wave owns one cout half (mt = wave&1) and keeps all 9 per-tap weight
// A-fragments in VGPRs (72 VGPRs).  The WMMA chain runs unconditionally
// (EXEC all-ones; all src LDS rows are defined, padding rows are zeros) and
// only the final store selects zeros for SAME-padding dst rows, so the loop
// body is straight-line and per-wave stalls are hidden by 4 waves/SIMD.
__device__ __forceinline__ void conv_layer_wmma(
    const _Float16* __restrict__ src,    // [nrows+2][130][32] (f-halo zeroed)
    _Float16*       __restrict__ dst,    // [nrows][130][32]
    const _Float16* __restrict__ wfrag,  // [9][2][32][16] pre-swizzled A frags
    const float*    __restrict__ bias,   // [32]
    int nrows, int tbase, int lane, int wave)
{
  const int mt     = wave & 1;           // cout half owned by this wave
  const int wslice = wave >> 1;          // 0..7: slice of (row, ntile) space
  const int m_hi   = lane >> 4;
  const int nn     = lane & 15;

  v16h aw[9];
  #pragma unroll
  for (int tap = 0; tap < 9; ++tap)      // loop-invariant weight fragments
    aw[tap] = *(const v16h*)&wfrag[((tap*2 + mt)*32 + lane)*16];

  float bias8[8];
  #pragma unroll
  for (int r = 0; r < 8; ++r) bias8[r] = bias[mt*16 + m_hi*8 + r];

  const int ntiles = nrows * 8;          // 8 ntiles of 16 f-positions per row
  #pragma unroll 2
  for (int tile = wslice; tile < ntiles; tile += 8) {
    int nt = tile & 7;
    int rr = tile >> 3;                  // dst row; t = tbase + rr
    int t  = tbase + rr;
    _Float16* d = &dst[(rr*130 + (1 + nt*16 + nn))*32 + mt*16 + m_hi*8];

    v8f c;
    #pragma unroll
    for (int r = 0; r < 8; ++r) c[r] = bias8[r];
    #pragma unroll
    for (int tap = 0; tap < 9; ++tap) {
      int kh = tap / 3, kw = tap % 3;
      // B 32x16: lanes 0-15 hold K=0..15, lanes 16-31 K=16..31 -> one v16h.
      v16h bb = *(const v16h*)&src[((rr + kh)*130 + (nt*16 + nn + kw))*32 + m_hi*16];
      c = __builtin_amdgcn_wmma_f32_16x16x32_f16(false, aw[tap], false, bb,
                                                 (short)0, c, false, false);
    }
    const bool inb = (t >= 0) && (t < T_);   // SAME padding row? (wave-uniform)
    v8h o;
    #pragma unroll
    for (int r = 0; r < 8; ++r) {
      float x = c[r] > 0.f ? c[r] : 0.f;     // ReLU
      o[r] = inb ? (_Float16)x : (_Float16)0.f;
    }
    *(v8h*)d = o;                            // one ds_store_b128
  }
}

__global__ __launch_bounds__(NTHREADS, 1)
void fused_tatum_conv(const float* __restrict__ pitch,   // (B, NT, F)
                      const float* __restrict__ onset,   // (B, NT)
                      const int*   __restrict__ tfrm,    // (B, NT+1)
                      const float* __restrict__ W1, const float* __restrict__ b1,
                      const float* __restrict__ W2, const float* __restrict__ b2,
                      const float* __restrict__ W3, const float* __restrict__ b3,
                      const float* __restrict__ W4, const float* __restrict__ b4,
                      float* __restrict__ out)           // (B, T, F)
{
  __shared__ __align__(64) char smem[SMEM_BYTES];
  __shared__ int   sIdx[12];
  __shared__ float sb1[32], sb2[32], sb3[32], sb4v[1];

  _Float16* sa1 = (_Float16*)(smem + OFF_A1);
  _Float16* sa2 = (_Float16*)(smem + OFF_A2);
  _Float16* sa3 = (_Float16*)(smem + OFF_A3);
  _Float16* sw2 = (_Float16*)(smem + OFF_W2);
  _Float16* sw3 = (_Float16*)(smem + OFF_W3);
  _Float16* sx0 = (_Float16*)(smem + OFF_X0);
  float*    sw1 = (float*)(smem + OFF_W1);
  float*    sw4 = (float*)(smem + OFF_W4);

  const int tid  = threadIdx.x;
  const int lane = tid & 31;
  const int wave = tid >> 5;
  const int bid  = blockIdx.x;
  const int b    = bid / (T_ / TILE);
  const int t0   = (bid % (T_ / TILE)) * TILE;

  // ---- Phase 0a: biases, weights, per-row tatum indices --------------------
  if (tid < 32) { sb1[tid] = b1[tid]; sb2[tid] = b2[tid]; sb3[tid] = b3[tid]; }
  if (tid == 0) sb4v[0] = b4[0];
  for (int i = tid; i < 32*2*9; i += NTHREADS) sw1[i] = W1[i];
  for (int i = tid; i < 32*9;   i += NTHREADS) sw4[i] = W4[i];

  // Pre-swizzle W2/W3 into this kernel's A-fragment (16x32) per-lane order (contiguous k, as the B loads):
  //   element e lives in VGPR e/2 (half e&1); K = hi*16 + e
  for (int u = tid; u < 9*2*32*16; u += NTHREADS) {
    int e   = u & 15;
    int ln  = (u >> 4) & 31;
    int mt  = (u >> 9) & 1;
    int tap = u >> 10;
    int m   = mt*16 + (ln & 15);
    int hi  = ln >> 4;
    int vg  = e >> 1;
    int k   = hi*16 + e; (void)vg;
    int kh  = tap / 3, kw = tap % 3;
    int widx = ((m*32 + k)*3 + kh)*3 + kw;   // (Cout, Cin, 3, 3)
    sw2[u] = (_Float16)W2[widx];
    sw3[u] = (_Float16)W3[widx];
  }

  if (tid < 12) {
    int tr = t0 - 4 + tid;
    int idx = -1;
    if (tr >= 0 && tr < T_) {
      const int* row = tfrm + b * (NT_ + 1);
      int lo = 1, hi = NT_ + 1;            // first j in [1,257) with row[j] > tr
      while (lo < hi) { int mid = (lo + hi) >> 1; if (row[mid] <= tr) lo = mid + 1; else hi = mid; }
      idx = lo - 1;                         // searchsorted(tf[1:], tr, 'right')
      if (idx > NT_ - 1) idx = NT_ - 1;
    }
    sIdx[tid] = idx;
  }
  __syncthreads();

  // ---- Phase 0b: tatum expansion -> x0 tile; zero f-halo cols of a1/a2 -----
  for (int u = tid; u < 12*130; u += NTHREADS) {
    int rr = u / 130, p = u % 130;
    int f = p - 1;
    int idx = sIdx[rr];
    float pr = 0.f, on = 0.f;
    if (idx >= 0 && f >= 0 && f < F_) {
      pr = pitch[(b*NT_ + idx)*F_ + f];
      on = onset[b*NT_ + idx];
    }
    sx0[u*2 + 0] = (_Float16)pr;
    sx0[u*2 + 1] = (_Float16)on;
  }
  for (int u = tid; u < 10*2*32; u += NTHREADS) {
    int rr = u >> 6, side = (u >> 5) & 1, c = u & 31;
    sa1[(rr*130 + side*129)*32 + c] = (_Float16)0.f;
    if (rr < 8) sa2[(rr*130 + side*129)*32 + c] = (_Float16)0.f;
  }
  __syncthreads();

  // ---- Phase 1: conv1 (Cin=2 -> 32, VALU) + ReLU -> a1 ---------------------
  for (int u = tid; u < 10*F_; u += NTHREADS) {
    int rr = u >> 7;                 // a1 row; t = t0-3+rr
    int f  = u & 127;
    int t  = t0 - 3 + rr;
    _Float16* dst = &sa1[(rr*130 + (f + 1))*32];
    if (t < 0 || t >= T_) {          // SAME padding: out-of-range rows are zero
      #pragma unroll
      for (int co = 0; co < 32; ++co) dst[co] = (_Float16)0.f;
      continue;
    }
    float acc[32];
    #pragma unroll
    for (int co = 0; co < 32; ++co) acc[co] = sb1[co];
    #pragma unroll
    for (int kh = 0; kh < 3; ++kh)
      #pragma unroll
      for (int kw = 0; kw < 3; ++kw)
        #pragma unroll
        for (int ch = 0; ch < 2; ++ch) {
          float v = (float)sx0[((rr + kh)*130 + (f + kw))*2 + ch];
          #pragma unroll
          for (int co = 0; co < 32; ++co)
            acc[co] += v * sw1[((co*2 + ch)*3 + kh)*3 + kw];
        }
    #pragma unroll
    for (int co = 0; co < 32; ++co) {
      float r = acc[co] > 0.f ? acc[co] : 0.f;
      dst[co] = (_Float16)r;
    }
  }
  __syncthreads();

  // ---- Phase 2: conv2 (32->32) WMMA + ReLU: a1 -> a2 -----------------------
  conv_layer_wmma(sa1, sa2, sw2, sb2, /*nrows=*/8, /*tbase=*/t0 - 2, lane, wave);
  __syncthreads();

  // ---- Phase 3: conv3 (32->32) WMMA + ReLU: a2 -> a3 (aliases dead a1) -----
  for (int u = tid; u < 6*2*32; u += NTHREADS) {       // zero a3 f-halo columns
    int rr = u >> 6, side = (u >> 5) & 1, c = u & 31;
    sa3[(rr*130 + side*129)*32 + c] = (_Float16)0.f;
  }
  conv_layer_wmma(sa2, sa3, sw3, sb3, /*nrows=*/6, /*tbase=*/t0 - 1, lane, wave);
  __syncthreads();

  // ---- Phase 4: conv4 (32 -> 1, VALU, no ReLU) -> global output ------------
  for (int u = tid; u < TILE*F_; u += NTHREADS) {
    int rr = u >> 7;                 // out row; t = t0+rr (always in range)
    int f  = u & 127;
    int t  = t0 + rr;
    float acc = sb4v[0];
    #pragma unroll
    for (int kh = 0; kh < 3; ++kh)
      #pragma unroll
      for (int kw = 0; kw < 3; ++kw) {
        const _Float16* src = &sa3[((rr + kh)*130 + (f + kw))*32];
        #pragma unroll
        for (int ci = 0; ci < 32; ++ci)
          acc += (float)src[ci] * sw4[(ci*3 + kh)*3 + kw];
      }
    out[(b*T_ + t)*F_ + f] = acc;
  }
}

extern "C" void kernel_launch(void* const* d_in, const int* in_sizes, int n_in,
                              void* d_out, int out_size, void* d_ws, size_t ws_size,
                              hipStream_t stream) {
  const float* pitch = (const float*)d_in[0];
  const float* onset = (const float*)d_in[1];
  const int*   tfrm  = (const int*)d_in[2];
  const float* W1 = (const float*)d_in[3]; const float* b1v = (const float*)d_in[4];
  const float* W2 = (const float*)d_in[5]; const float* b2v = (const float*)d_in[6];
  const float* W3 = (const float*)d_in[7]; const float* b3v = (const float*)d_in[8];
  const float* W4 = (const float*)d_in[9]; const float* b4v = (const float*)d_in[10];
  float* out = (float*)d_out;
  (void)in_sizes; (void)n_in; (void)out_size; (void)d_ws; (void)ws_size;

  dim3 grid(B_ * (T_ / TILE));   // 4096 blocks
  dim3 block(NTHREADS);          // 16 wave32 waves
  hipLaunchKernelGGL(fused_tatum_conv, grid, block, 0, stream,
                     pitch, onset, tfrm, W1, b1v, W2, b2v, W3, b3v, W4, b4v, out);
}
